// RNNStepPredictor_23029614641473
// MI455X (gfx1250) — matrix operand order repaired and checked by reading, not compiled, not run on hardware
//
#include <hip/hip_runtime.h>
#include <cstdint>

typedef _Float16 v16h __attribute__((ext_vector_type(16)));
typedef _Float16 v8h  __attribute__((ext_vector_type(8)));
typedef float    v8f  __attribute__((ext_vector_type(8)));

#define Bsz 512
#define Tn  256
#define Un  512
#define Fn  2
#define NG  2048           // 4*U
#define MT  16             // batch rows per workgroup (WMMA M)
#define THREADS 512        // 16 waves
#define UT_PER_WAVE 2      // (U/16) tiles / 16 waves

__global__ void convert_f32_to_f16(const float* __restrict__ src,
                                   _Float16* __restrict__ dst, int n) {
  int i = blockIdx.x * blockDim.x + threadIdx.x;
  if (i < n) dst[i] = (_Float16)src[i];
}

__device__ __forceinline__ float sigmoidf_fast(float x) {
  return 1.0f / (1.0f + __expf(-x));
}

__global__ __launch_bounds__(THREADS)
void lstm_warmup_kernel(const float* __restrict__ X,      // [B,T,F]
                        const float* __restrict__ Wx,     // [F,4U]
                        const _Float16* __restrict__ WhH, // [U,4U] f16 (d_ws)
                        const float* __restrict__ bvec,   // [4U]
                        const float* __restrict__ Wd,     // [U,F]
                        const float* __restrict__ bd,     // [F]
                        float* __restrict__ out)          // preds|lstm|hT|cT
{
  __shared__ __align__(16) _Float16 h16[MT][Un + 8]; // padded: bank-conflict free
  __shared__ float h32[MT][Un + 4];
  __shared__ float xbuf[MT][Fn];
  __shared__ float predscr[32][16];

  const int tid  = threadIdx.x;
  const int lane = tid & 31;
  const int wv   = tid >> 5;
  const int l15  = lane & 15;
  const int hi8  = (lane >> 4) * 8;           // A-layout K-offset == C-layout M-offset
  const int m0   = blockIdx.x * MT;
  const int ubase = wv * (UT_PER_WAVE * 16);  // this wave's gate-column base in [0,512)

  const size_t PRED_OFF = 0;
  const size_t LSTM_OFF = (size_t)Bsz * Tn * Fn;
  const size_t HT_OFF   = LSTM_OFF + (size_t)Bsz * Tn * Un;
  const size_t CT_OFF   = HT_OFF + (size_t)Bsz * Un;

  // h0 = 0
  for (int idx = tid; idx < MT * Un; idx += THREADS) {
    int r = idx >> 9, c = idx & (Un - 1);
    h16[r][c] = (_Float16)0.0f;
    h32[r][c] = 0.0f;
  }

  v8f acc[UT_PER_WAVE][4];   // zi,zf,zg,zo accumulators per u-tile
  v8f cst[UT_PER_WAVE];      // cell state, lives in VGPRs all 256 steps
  #pragma unroll
  for (int u = 0; u < UT_PER_WAVE; ++u)
    #pragma unroll
    for (int v = 0; v < 8; ++v) cst[u][v] = 0.0f;

  __syncthreads();

  #pragma unroll 1
  for (int t = 0; t < Tn; ++t) {
    // ---- stage 1: stage x_t into LDS ----
    if (tid < MT * Fn) {
      int m = tid >> 1, f = tid & 1;
      xbuf[m][f] = X[((size_t)(m0 + m) * Tn + t) * Fn + f];
    }
    __syncthreads(); // A: x visible; prev-step h16 stable for reading

    float xr0[8], xr1[8];
    #pragma unroll
    for (int v = 0; v < 8; ++v) {
      int m = v + hi8;
      xr0[v] = xbuf[m][0];
      xr1[v] = xbuf[m][1];
    }

    // ---- stage 2: acc = x@Wx + b (scalar, F=2), then z += h@Wh via WMMA ----
    #pragma unroll
    for (int u = 0; u < UT_PER_WAVE; ++u) {
      #pragma unroll
      for (int g = 0; g < 4; ++g) {
        int n = g * Un + ubase + u * 16 + l15;   // C-layout: N = lane&15
        float w0 = Wx[n], w1 = Wx[NG + n], bb = bvec[n];
        #pragma unroll
        for (int v = 0; v < 8; ++v)              // C-layout: M = v + 8*(lane>>4)
          acc[u][g][v] = fmaf(xr0[v], w0, fmaf(xr1[v], w1, bb));
      }
    }

    for (int kk = 0; kk < Un; kk += 32) {
      // A fragment: lane row M=l15, K = kk + {hi8..hi8+7, 16+hi8..16+hi8+7}
      const _Float16* hrow = &h16[l15][0];
      v8h a0 = *(const v8h*)(hrow + kk + hi8);
      v8h a1 = *(const v8h*)(hrow + kk + 16 + hi8);
      v16h afrag;
      #pragma unroll
      for (int i = 0; i < 8; ++i) { afrag[i] = a0[i]; afrag[i + 8] = a1[i]; }

      // B fragments: lane column N=l15, K = kk + {hi8..hi8+7, 16+hi8..16+hi8+7}
      const _Float16* wrow = WhH + (size_t)kk * NG + l15;
      #pragma unroll
      for (int u = 0; u < UT_PER_WAVE; ++u) {
        #pragma unroll
        for (int g = 0; g < 4; ++g) {
          int nb = g * Un + ubase + u * 16;
          v16h bfrag;
          #pragma unroll
          for (int i = 0; i < 8; ++i) { bfrag[i] = wrow[(size_t)(hi8 + i) * NG + nb]; bfrag[i + 8] = wrow[(size_t)(16 + hi8 + i) * NG + nb]; }
          acc[u][g] = __builtin_amdgcn_wmma_f32_16x16x32_f16(
              false, afrag, false, bfrag, (short)0, acc[u][g], false, false);
        }
      }
    }
    __syncthreads(); // B: all waves done reading h16 of step t-1

    // ---- stage 3: gates, c/h update, write h to LDS (f16 + f32) ----
    #pragma unroll
    for (int u = 0; u < UT_PER_WAVE; ++u) {
      int col = ubase + u * 16 + l15;
      #pragma unroll
      for (int v = 0; v < 8; ++v) {
        int m = v + hi8;
        float ig = sigmoidf_fast(acc[u][0][v]);
        float fg = sigmoidf_fast(acc[u][1][v]);
        float gg = tanhf(acc[u][2][v]);
        float og = sigmoidf_fast(acc[u][3][v]);
        float cn = fg * cst[u][v] + ig * gg;
        cst[u][v] = cn;
        float hn = og * tanhf(cn);
        h16[m][col] = (_Float16)hn;
        h32[m][col] = hn;
      }
    }
    __syncthreads(); // C: new h visible

    // ---- stage 4: stream h to lstm_outputs; prediction-head partials ----
    for (int idx = tid; idx < MT * Un; idx += THREADS) {
      int r = idx >> 9, c = idx & (Un - 1);
      out[LSTM_OFF + ((size_t)(m0 + r) * Tn + t) * Un + c] = h32[r][c];
    }
    {
      int g = tid >> 4, s = tid & 15;   // 32 (m,f) outputs x 16-thread reductions
      int m = g >> 1, f = g & 1;
      float p = 0.0f;
      for (int k = s * 32; k < s * 32 + 32; ++k)
        p = fmaf(h32[m][k], Wd[k * Fn + f], p);
      predscr[g][s] = p;
    }
    __syncthreads(); // D: partials visible

    if (tid < 32) {
      int m = tid >> 1, f = tid & 1;
      float p = bd[f];
      #pragma unroll
      for (int s = 0; s < 16; ++s) p += predscr[tid][s];
      out[PRED_OFF + ((size_t)(m0 + m) * Tn + t) * Fn + f] = tanhf(p);
    }
  }

  __syncthreads();
  // hT from final LDS copy (coalesced)
  for (int idx = tid; idx < MT * Un; idx += THREADS) {
    int r = idx >> 9, c = idx & (Un - 1);
    out[HT_OFF + (size_t)(m0 + r) * Un + c] = h32[r][c];
  }
  // cT straight from registers (one-time scattered stores)
  #pragma unroll
  for (int u = 0; u < UT_PER_WAVE; ++u) {
    int col = ubase + u * 16 + l15;
    #pragma unroll
    for (int v = 0; v < 8; ++v)
      out[CT_OFF + (size_t)(m0 + v + hi8) * Un + col] = cst[u][v];
  }
}

extern "C" void kernel_launch(void* const* d_in, const int* in_sizes, int n_in,
                              void* d_out, int out_size, void* d_ws, size_t ws_size,
                              hipStream_t stream) {
  const float* X  = (const float*)d_in[0];   // start_features [512,256,2]
  const float* Wx = (const float*)d_in[1];   // [2,2048]
  const float* Wh = (const float*)d_in[2];   // [512,2048]
  const float* b  = (const float*)d_in[3];   // [2048]
  const float* Wd = (const float*)d_in[4];   // [512,2]
  const float* bd = (const float*)d_in[5];   // [2]
  float* out = (float*)d_out;
  _Float16* WhH = (_Float16*)d_ws;           // 2 MB f16 copy of Wh

  int n = Un * NG;
  convert_f32_to_f16<<<(n + 255) / 256, 256, 0, stream>>>(Wh, WhH, n);
  lstm_warmup_kernel<<<dim3(Bsz / MT), dim3(THREADS), 0, stream>>>(
      X, Wx, WhH, b, Wd, bd, out);
}
